// SO3Activation_17678085390535
// MI455X (gfx1250) — matrix operand order repaired and checked by reading, not compiled, not run on hardware
//
#include <hip/hip_runtime.h>

typedef _Float16 v16h __attribute__((ext_vector_type(16)));
typedef _Float16 v8h  __attribute__((ext_vector_type(8)));
typedef float    v8f  __attribute__((ext_vector_type(8)));

#define DIM   165
#define KP    192          // padded K (6 x 32)
#define G_TOT 87808        // 56*28*56
#define M_TOT 512          // 32*16
#define NCH   (G_TOT / 32) // 2744 chunks of 32 g-columns
#define NSPLIT 128

// ---------------- prep kernels ----------------

__global__ void so3_prep_x(const float* __restrict__ x, _Float16* __restrict__ Xh) {
    int i = blockIdx.x * blockDim.x + threadIdx.x;
    if (i >= M_TOT * KP) return;
    int m = i / KP, k = i % KP;
    float v = (k < DIM) ? x[m * DIM + k] : 0.0f;
    Xh[i] = (_Float16)v;
}

__global__ void so3_prep_d(const float* __restrict__ D,
                           _Float16* __restrict__ Drh,   // [G, KP] row-major
                           _Float16* __restrict__ Dth) { // [KP, G] row-major
    size_t i = (size_t)blockIdx.x * blockDim.x + threadIdx.x;
    if (i >= (size_t)G_TOT * KP) return;
    size_t g = i / KP, k = i % KP;
    float v = (k < DIM) ? D[g * DIM + k] : 0.0f;
    _Float16 h = (_Float16)v;
    Drh[i] = h;
    Dth[k * (size_t)G_TOT + g] = h;
}

__global__ void so3_zero_out(float* __restrict__ out) {
    int i = blockIdx.x * blockDim.x + threadIdx.x;
    if (i < M_TOT * DIM) out[i] = 0.0f;
}

// ---------------- fused WMMA kernel ----------------
// One wave (32 threads) per block. blockIdx.y = M-tile (16 rows of x),
// blockIdx.x = G split; chunks of 32 g-columns are strided across splits.

__global__ __launch_bounds__(32)
void so3_fused(const float* __restrict__ qw,
               const _Float16* __restrict__ Xh,
               const _Float16* __restrict__ Drh,
               const _Float16* __restrict__ Dth,
               float* __restrict__ out) {
    __shared__ __align__(32) _Float16 lds_g[16 * 32]; // g tile relayout buffer

    const int lane = threadIdx.x;   // 0..31
    const int hi   = lane >> 4;     // 0/1: lane half selects K/M sub-block
    const int lo   = lane & 15;
    const int mt   = blockIdx.y;    // 0..31

    // ---- A operand for GEMM1: x tile 16x192 as 6 x (16x32) in WMMA A layout ----
    // A order used here (contiguous k, same as B): lane row M=lo; halves 0..15 = K klo..klo+15,
    // klo = hi*16.
    v16h axt[6];
    {
        const _Float16* xrow = Xh + (size_t)(mt * 16 + lo) * KP;
#pragma unroll
        for (int kc = 0; kc < 6; ++kc) {
            const _Float16* p = xrow + kc * 32 + hi * 16;
            v8h l0 = *(const v8h*)p;
            v8h l1 = *(const v8h*)(p + 8);
            v16h a;
#pragma unroll
            for (int i = 0; i < 8; ++i) { a[i] = l0[i]; a[i + 8] = l1[i]; }
            axt[kc] = a;
        }
    }

    // ---- output tile accumulators: 16 x 192 = 12 WMMA C tiles ----
    v8f acc[12];
#pragma unroll
    for (int nt = 0; nt < 12; ++nt) { v8f z = {}; acc[nt] = z; }

    for (int ch = blockIdx.x; ch < NCH; ch += gridDim.x) {
        const int g0 = ch * 32;

        // ---- GEMM1: g(16x32) = x(16x192) @ D^T(192x32), fp32 accum ----
        // B layout: lane col N=lo, halves j: K = hi*16 + j -> 16 contiguous
        // halves of row (g0 [+16] + lo) of Drh. One 32B load per lane.
        v8f c0 = {}, c1 = {};
#pragma unroll
        for (int kc = 0; kc < 6; ++kc) {
            const _Float16* pb0 = Drh + (size_t)(g0 + lo) * KP + kc * 32 + hi * 16;
            const _Float16* pb1 = pb0 + (size_t)16 * KP;
            v16h b0 = *(const v16h*)pb0;
            v16h b1 = *(const v16h*)pb1;
            c0 = __builtin_amdgcn_wmma_f32_16x16x32_f16(false, axt[kc], false, b0,
                                                        (short)0, c0, false, false);
            c1 = __builtin_amdgcn_wmma_f32_16x16x32_f16(false, axt[kc], false, b1,
                                                        (short)0, c1, false, false);
        }

        // prefetch next chunk's D rows while we do the relayout
        {
            int chn = ch + gridDim.x;
            if (chn < NCH) {
                __builtin_prefetch(Drh + (size_t)(chn * 32 + lane) * KP, 0, 1);
            }
        }

        // ---- relu + quadrature weight (per g-column), write f16 to LDS ----
        // C layout: lane col N=lo, VGPR r -> row M = r + 8*hi.
        const float q0 = qw[g0 + lo];
        const float q1 = qw[g0 + 16 + lo];
#pragma unroll
        for (int r = 0; r < 8; ++r) {
            float v0 = c0[r]; v0 = v0 > 0.0f ? v0 : 0.0f;
            float v1 = c1[r]; v1 = v1 > 0.0f ? v1 : 0.0f;
            const int row = r + 8 * hi;
            lds_g[row * 32 + lo]      = (_Float16)(v0 * q0);
            lds_g[row * 32 + 16 + lo] = (_Float16)(v1 * q1);
        }
        // cross-lane LDS handoff within the wave: order + completion
        asm volatile("s_wait_dscnt 0" ::: "memory");

        // ---- reload g tile in WMMA A layout (16x32) ----
        v16h ag;
        {
            const _Float16* p = &lds_g[lo * 32 + hi * 16];
            v8h l0 = *(const v8h*)p;
            v8h l1 = *(const v8h*)(p + 8);
#pragma unroll
            for (int i = 0; i < 8; ++i) { ag[i] = l0[i]; ag[i + 8] = l1[i]; }
        }
        asm volatile("s_wait_dscnt 0" ::: "memory");

        // ---- GEMM2: out(16x192) += g(16x32) @ D(32x192) ----
        // B layout: lane col N = nt*16+lo (dim), halves j: K = hi*16 + j (g)
        // -> 16 contiguous halves of Dth row (nt*16+lo). One 32B load per lane.
#pragma unroll
        for (int nt = 0; nt < 12; ++nt) {
            const _Float16* pb = Dth + (size_t)(nt * 16 + lo) * G_TOT + g0 + hi * 16;
            v16h b = *(const v16h*)pb;
            acc[nt] = __builtin_amdgcn_wmma_f32_16x16x32_f16(false, ag, false, b,
                                                             (short)0, acc[nt], false, false);
        }
    }

    // ---- accumulate partial output (per-split) into d_out ----
#pragma unroll
    for (int nt = 0; nt < 12; ++nt) {
        const int dimc = nt * 16 + lo;
        if (dimc < DIM) {
#pragma unroll
            for (int r = 0; r < 8; ++r) {
                atomicAdd(&out[(size_t)(mt * 16 + r + 8 * hi) * DIM + dimc], acc[nt][r]);
            }
        }
    }
}

// ---------------- host side ----------------

extern "C" void kernel_launch(void* const* d_in, const int* in_sizes, int n_in,
                              void* d_out, int out_size, void* d_ws, size_t ws_size,
                              hipStream_t stream) {
    const float* x  = (const float*)d_in[0];   // [512, 165]
    const float* D  = (const float*)d_in[1];   // [87808, 165]
    const float* qw = (const float*)d_in[2];   // [87808]
    float* out = (float*)d_out;                // [512, 165]

    _Float16* Xh  = (_Float16*)d_ws;                       // 512*192 halves
    _Float16* Drh = Xh + (size_t)M_TOT * KP;               // G*192 halves
    _Float16* Dth = Drh + (size_t)G_TOT * KP;              // 192*G halves
    // total ws use: ~67.6 MB

    {
        int n = M_TOT * KP;
        so3_prep_x<<<(n + 255) / 256, 256, 0, stream>>>(x, Xh);
    }
    {
        size_t n = (size_t)G_TOT * KP;
        so3_prep_d<<<(unsigned)((n + 255) / 256), 256, 0, stream>>>(D, Drh, Dth);
    }
    {
        int n = M_TOT * DIM;
        so3_zero_out<<<(n + 255) / 256, 256, 0, stream>>>(out);
    }
    {
        dim3 grid(NSPLIT, M_TOT / 16); // 128 G-splits x 32 M-tiles
        so3_fused<<<grid, 32, 0, stream>>>(qw, Xh, Drh, Dth, out);
    }
}
